// ContactGraspNet_60017872994740
// MI455X (gfx1250) — matrix operand order repaired and checked by reading, not compiled, not run on hardware
//
#include <hip/hip_runtime.h>

typedef __attribute__((ext_vector_type(16))) _Float16 v16h;
typedef __attribute__((ext_vector_type(8)))  _Float16 v8h;
typedef __attribute__((ext_vector_type(8)))  float    v8f;

#define BN_INV 0.999995000037499f  /* 1/sqrt(1+1e-5) */

__constant__ float BIN_VALS_D[10] = {
    0.003972176645f, 0.0119165277f, 0.0198608765f, 0.0278052253f, 0.0357495740f,
    0.0436939228f,   0.0516382716f, 0.0595826204f, 0.0675269692f, 0.075f};

// ---- CDNA5 async global->LDS (ASYNCcnt path, ISA 15.18.3 / 08_async_tensor) ----
__device__ __forceinline__ unsigned ldsOff(const void* p) {
  // generic pointers into the LDS aperture carry the LDS byte offset in addr[31:0]
  return (unsigned)(unsigned long long)p;
}
__device__ __forceinline__ void async_ld_b128(unsigned lds_byte_off, const void* gptr) {
  asm volatile("global_load_async_to_lds_b128 %0, %1, off"
               :: "v"(lds_byte_off), "v"(gptr)
               : "memory");
}
__device__ __forceinline__ void wait_async0() {
  asm volatile("s_wait_asynccnt 0x0" ::: "memory");
}

// ---------------------------------------------------------------- utilities
// fp32 [cin,cout] -> fp16 TRANSPOSED [cout, cinPad] (zero-padded K)
__global__ void k_w2h(const float* __restrict__ W, _Float16* __restrict__ Wt,
                      int cin, int cinPad, int cout) {
  int t = blockIdx.x * blockDim.x + threadIdx.x;
  int total = cinPad * cout;
  if (t >= total) return;
  int c = t / cinPad, r = t % cinPad;
  Wt[t] = (r < cin) ? (_Float16)W[r * cout + c] : (_Float16)0.0f;
}

// ---------------------------------------------------------------- FPS
__global__ void k_fps(const float* __restrict__ xyz, float* __restrict__ dist,
                      int* __restrict__ fidx, int N, int npoint) {
  int b = blockIdx.x;
  const float* p = xyz + (long long)b * N * 3;
  float* d = dist + (long long)b * N;
  __shared__ float sval[256];
  __shared__ int   sidx[256];
  __shared__ float cx, cy, cz;
  int far = 0;
  for (int i = threadIdx.x; i < N; i += blockDim.x) d[i] = 1e10f;
  __syncthreads();
  for (int it = 0; it < npoint; ++it) {
    if (threadIdx.x == 0) {
      fidx[b * npoint + it] = far;
      cx = p[far * 3 + 0]; cy = p[far * 3 + 1]; cz = p[far * 3 + 2];
    }
    __syncthreads();
    float best = -1.f; int bi = 0;
    for (int i = threadIdx.x; i < N; i += blockDim.x) {
      float dx = p[i * 3 + 0] - cx, dy = p[i * 3 + 1] - cy, dz = p[i * 3 + 2] - cz;
      float dd = dx * dx + dy * dy + dz * dz;
      float m = fminf(d[i], dd);
      d[i] = m;
      if (m > best) { best = m; bi = i; }
    }
    sval[threadIdx.x] = best; sidx[threadIdx.x] = bi;
    __syncthreads();
    for (int s = 128; s > 0; s >>= 1) {
      if ((int)threadIdx.x < s) {
        float ov = sval[threadIdx.x + s]; int oi = sidx[threadIdx.x + s];
        if (ov > sval[threadIdx.x] ||
            (ov == sval[threadIdx.x] && oi < sidx[threadIdx.x])) {
          sval[threadIdx.x] = ov; sidx[threadIdx.x] = oi;
        }
      }
      __syncthreads();
    }
    far = sidx[0];
    __syncthreads();
  }
}

__global__ void k_gather3(const float* __restrict__ xyz, const int* __restrict__ idx,
                          float* __restrict__ out, int Bc, int S, int N) {
  int t = blockIdx.x * blockDim.x + threadIdx.x;
  if (t >= Bc * S) return;
  int b = t / S;
  int i = idx[t];
  const float* src = xyz + ((long long)b * N + i) * 3;
  out[t * 3 + 0] = src[0]; out[t * 3 + 1] = src[1]; out[t * 3 + 2] = src[2];
}

// ---------------------------------------------------------------- ball query
__global__ void k_ballquery(const float* __restrict__ nxyz, const float* __restrict__ xyz,
                            int* __restrict__ gidx, int Bc, int S, int N, float r2, int K) {
  int t = blockIdx.x * blockDim.x + threadIdx.x;
  if (t >= Bc * S) return;
  int b = t / S;
  const float* c = nxyz + (long long)t * 3;
  const float* p = xyz + (long long)b * N * 3;
  int* out = gidx + (long long)t * K;
  int cnt = 0;
  for (int i = 0; i < N && cnt < K; ++i) {
    float dx = p[i * 3 + 0] - c[0], dy = p[i * 3 + 1] - c[1], dz = p[i * 3 + 2] - c[2];
    if (dx * dx + dy * dy + dz * dz <= r2) out[cnt++] = i;
  }
  int pad = (cnt > 0) ? out[0] : 0;
  for (; cnt < K; ++cnt) out[cnt] = pad;
}

// ---------------------------------------------------------------- grouping (feats ++ rel_xyz), fp16, zero-pad
__global__ void k_group(const float* __restrict__ xyz, const float* __restrict__ nxyz,
                        const float* __restrict__ feats, const int* __restrict__ gidx,
                        _Float16* __restrict__ X, int Bc, int S, int N, int K,
                        int D, int cinPad) {
  long long t = (long long)blockIdx.x * blockDim.x + threadIdx.x;
  long long total = (long long)Bc * S * K;
  if (t >= total) return;
  long long bs = t / K;
  int s = (int)(bs % S);
  int b = (int)(bs / S);
  int idx = gidx[t];
  if (idx >= N) idx = N - 1;
  _Float16* Xr = X + t * cinPad;
  const float* fsrc = feats + ((long long)b * N + idx) * D;
  for (int d = 0; d < D; ++d) Xr[d] = (_Float16)fsrc[d];
  const float* p = xyz + ((long long)b * N + idx) * 3;
  const float* c = nxyz + ((long long)b * S + s) * 3;
  Xr[D + 0] = (_Float16)(p[0] - c[0]);
  Xr[D + 1] = (_Float16)(p[1] - c[1]);
  Xr[D + 2] = (_Float16)(p[2] - c[2]);
  for (int d = D + 3; d < cinPad; ++d) Xr[d] = (_Float16)0.0f;
}

// ---------------------------------------------------------------- WMMA GEMM + bias + eval-BN + ReLU
// Block = 8 waves; block tile = 128(M) x 32(N); each wave: one A frag, two 16x16 WMMAs.
// A/B tiles staged via GLOBAL_LOAD_ASYNC_TO_LDS_B128 (ASYNCcnt), weights pre-transposed [cout, Kd].
__global__ __launch_bounds__(256) void k_wmma_mlp(
    const _Float16* __restrict__ X, const _Float16* __restrict__ Wt,
    const float* __restrict__ bias, const float* __restrict__ gamma,
    const float* __restrict__ beta, _Float16* __restrict__ Yh,
    float* __restrict__ Yf, int M, int Kd, int N) {
  __shared__ __align__(16) _Float16 As[128 * 32];   // [m][k]
  __shared__ __align__(16) _Float16 Bst[32 * 32];   // [n][k] (transposed)
  const int wave = threadIdx.x >> 5;
  const int lane = threadIdx.x & 31;
  const int tileM = blockIdx.x * 128 + wave * 16;
  const int tileN = blockIdx.y * 32;
  const unsigned baseA = ldsOff(As);
  const unsigned baseB = ldsOff(Bst);
  v8f acc0 = {0.f, 0.f, 0.f, 0.f, 0.f, 0.f, 0.f, 0.f};
  v8f acc1 = {0.f, 0.f, 0.f, 0.f, 0.f, 0.f, 0.f, 0.f};

  for (int k0 = 0; k0 < Kd; k0 += 32) {
    // ---- stage A: 128x32 halves = 512 x 16B chunks, 2 per thread (async DMA to LDS)
#pragma unroll
    for (int qq = 0; qq < 2; ++qq) {
      int q = (int)threadIdx.x + qq * 256;
      int r = q >> 2;
      int j = (q & 3) << 3;
      int gr = blockIdx.x * 128 + r;
      if (gr >= M) gr = M - 1;                     // clamp; unstored rows only
      async_ld_b128(baseA + (unsigned)(r * 32 + j) * 2,
                    X + (long long)gr * Kd + k0 + j);
    }
    // ---- stage B: 32x32 halves = 128 x 16B chunks (waves 0-3)
    if (threadIdx.x < 128) {
      int n = (int)threadIdx.x >> 2;
      int j = ((int)threadIdx.x & 3) << 3;
      async_ld_b128(baseB + (unsigned)(n * 32 + j) * 2,
                    Wt + (long long)(tileN + n) * Kd + k0 + j);
    }
    // gfx1250 prefetch of next K-slab
    if (k0 + 32 < Kd && threadIdx.x < 128) {
      int gr = blockIdx.x * 128 + (int)threadIdx.x;
      if (gr < M) __builtin_prefetch(&X[(long long)gr * Kd + k0 + 32], 0, 1);
    }
    wait_async0();
    __syncthreads();

    // ---- A fragment (ISA 7.12.2): lanes 0-15 K{0..7,16..23}, lanes 16-31 K{8..15,24..31}
    const int row = lane & 15;
    const int kh = (lane >> 4) << 3;
    v8h alo = *(const v8h*)(As + (wave * 16 + row) * 32 + kh);
    v8h ahi = *(const v8h*)(As + (wave * 16 + row) * 32 + 16 + kh);
    v16h a = {};
#pragma unroll
    for (int i = 0; i < 8; ++i) { a[i] = alo[i]; a[8 + i] = ahi[i]; }

    // ---- B fragments from [n][k] LDS: lane n = lane&15 (+16 for 2nd tile), lanes 0-15 K{0..7,16..23}, lanes 16-31 K{8..15,24..31}
    const int nb = lane & 15;
    const int kb = (lane >> 4) << 3;
    v8h b0lo = *(const v8h*)(Bst + nb * 32 + kb);
    v8h b0hi = *(const v8h*)(Bst + nb * 32 + kb + 16);
    v8h b1lo = *(const v8h*)(Bst + (nb + 16) * 32 + kb);
    v8h b1hi = *(const v8h*)(Bst + (nb + 16) * 32 + kb + 16);
    v16h b0 = {}, b1 = {};
#pragma unroll
    for (int i = 0; i < 8; ++i) {
      b0[i] = b0lo[i]; b0[8 + i] = b0hi[i];
      b1[i] = b1lo[i]; b1[8 + i] = b1hi[i];
    }
    acc0 = __builtin_amdgcn_wmma_f32_16x16x32_f16(false, a, false, b0,
                                                  (short)0, acc0, false, false);
    acc1 = __builtin_amdgcn_wmma_f32_16x16x32_f16(false, a, false, b1,
                                                  (short)0, acc1, false, false);
    __syncthreads();
  }

  // ---- epilogue: C/D layout N = lane&15, M = (lane>>4)*8 + vgpr
  const int nb = lane & 15;
  const int mrow = (lane >> 4) << 3;
#pragma unroll
  for (int h = 0; h < 2; ++h) {
    int col = tileN + nb + 16 * h;
    float bsv = bias ? bias[col] : 0.f;
    float sc = gamma ? (BN_INV * gamma[col]) : 1.f;
    float sh = beta ? beta[col] : 0.f;
#pragma unroll
    for (int r = 0; r < 8; ++r) {
      int gr = tileM + mrow + r;
      if (gr < M) {
        float v = (h == 0) ? acc0[r] : acc1[r];
        v = fmaxf((v + bsv) * sc + sh, 0.f);
        if (Yh) Yh[(long long)gr * N + col] = (_Float16)v;
        if (Yf) Yf[(long long)gr * N + col] = v;
      }
    }
  }
}

// ---------------------------------------------------------------- max over K
__global__ void k_maxpool(const _Float16* __restrict__ Y, float* __restrict__ out,
                          int Bc, int S, int K, int C, int Ctot, int coff) {
  long long t = (long long)blockIdx.x * blockDim.x + threadIdx.x;
  long long total = (long long)Bc * S * C;
  if (t >= total) return;
  int c = (int)(t % C);
  long long bs = t / C;
  const _Float16* y = Y + bs * (long long)K * C + c;
  float m = -3.4e38f;
  for (int k = 0; k < K; ++k) m = fmaxf(m, (float)y[(long long)k * C]);
  out[bs * Ctot + coff + c] = m;
}

// ---------------------------------------------------------------- 3-NN interpolation
__global__ void k_interp3(const float* __restrict__ xyz1, const float* __restrict__ xyz2,
                          const float* __restrict__ p2, float* __restrict__ out,
                          int Bc, int N, int S, int C) {
  int t = blockIdx.x * blockDim.x + threadIdx.x;
  if (t >= Bc * N) return;
  int n = t % N, b = t / N;
  const float* q = xyz1 + ((long long)b * N + n) * 3;
  const float* ref = xyz2 + (long long)b * S * 3;
  float d0 = 3e38f, d1 = 3e38f, d2 = 3e38f; int i0 = 0, i1 = 0, i2 = 0;
  for (int s = 0; s < S; ++s) {
    float dx = ref[s * 3 + 0] - q[0], dy = ref[s * 3 + 1] - q[1], dz = ref[s * 3 + 2] - q[2];
    float d = dx * dx + dy * dy + dz * dz;
    if (d < d0)      { d2 = d1; i2 = i1; d1 = d0; i1 = i0; d0 = d; i0 = s; }
    else if (d < d1) { d2 = d1; i2 = i1; d1 = d; i1 = s; }
    else if (d < d2) { d2 = d; i2 = s; }
  }
  float w0 = 1.f / (d0 + 1e-8f), w1 = 1.f / (d1 + 1e-8f), w2 = 1.f / (d2 + 1e-8f);
  float ws = w0 + w1 + w2; w0 /= ws; w1 /= ws; w2 /= ws;
  const float* f = p2 + (long long)b * S * C;
  float* o = out + ((long long)b * N + n) * C;
  for (int c = 0; c < C; ++c)
    o[c] = w0 * f[(long long)i0 * C + c] + w1 * f[(long long)i1 * C + c] +
           w2 * f[(long long)i2 * C + c];
}

__global__ void k_broadcast(const float* __restrict__ p2, float* __restrict__ out,
                            int Bc, int N, int C) {
  long long t = (long long)blockIdx.x * blockDim.x + threadIdx.x;
  long long total = (long long)Bc * N * C;
  if (t >= total) return;
  int c = (int)(t % C);
  int b = (int)(t / ((long long)N * C));
  out[t] = p2[(long long)b * C + c];
}

__global__ void k_concat2h(const float* __restrict__ A, const float* __restrict__ Bf,
                           _Float16* __restrict__ X, long long rows, int C1, int C2,
                           int cinPad) {
  long long t = (long long)blockIdx.x * blockDim.x + threadIdx.x;
  long long total = rows * cinPad;
  if (t >= total) return;
  int c = (int)(t % cinPad);
  long long r = t / cinPad;
  float v = 0.f;
  if (c < C1) v = A[r * C1 + c];
  else if (c < C1 + C2) v = Bf[r * C2 + (c - C1)];
  X[t] = (_Float16)v;
}

// ---------------------------------------------------------------- small final linear (cout 1..10)
__global__ void k_linear2(const _Float16* __restrict__ H, const float* __restrict__ W,
                          const float* __restrict__ b, float* __restrict__ out,
                          int M, int cout) {
  int t = blockIdx.x * blockDim.x + threadIdx.x;
  if (t >= M * cout) return;
  int c = t % cout, m = t / cout;
  const _Float16* h = H + (long long)m * 128;
  float s = b[c];
  for (int i = 0; i < 128; ++i) s += (float)h[i] * W[i * cout + c];
  out[(long long)m * cout + c] = s;
}

// ---------------------------------------------------------------- grasp assembly / sigmoid / xyz passthrough
__global__ void k_post(const float* __restrict__ gdr, const float* __restrict__ adr,
                       const float* __restrict__ wlog, const float* __restrict__ segr,
                       const float* __restrict__ nxyz1, float* __restrict__ out,
                       int Bc, int N) {
  int t = blockIdx.x * blockDim.x + threadIdx.x;
  long long NT = (long long)Bc * N;
  if (t >= NT) return;
  float gx = gdr[t * 3 + 0], gy = gdr[t * 3 + 1], gz = gdr[t * 3 + 2];
  float gn = fmaxf(sqrtf(gx * gx + gy * gy + gz * gz), 1e-12f);
  gx /= gn; gy /= gn; gz /= gn;
  float ax = adr[t * 3 + 0], ay = adr[t * 3 + 1], az = adr[t * 3 + 2];
  float dot = ax * gx + ay * gy + az * gz;
  ax -= dot * gx; ay -= dot * gy; az -= dot * gz;
  float an = fmaxf(sqrtf(ax * ax + ay * ay + az * az), 1e-12f);
  ax /= an; ay /= an; az /= an;
  const float* wl = wlog + (long long)t * 10;
  int am = 0; float bv = wl[0];
  for (int i = 1; i < 10; ++i) if (wl[i] > bv) { bv = wl[i]; am = i; }
  float width = BIN_VALS_D[am];
  float cx = ay * gz - az * gy, cy = az * gx - ax * gz, cz = ax * gy - ay * gx;
  const float* pt = nxyz1 + (long long)t * 3;
  float tx = pt[0] + 0.5f * width * gx - 0.1034f * ax;
  float ty = pt[1] + 0.5f * width * gy - 0.1034f * ay;
  float tz = pt[2] + 0.5f * width * gz - 0.1034f * az;
  float* G = out + (long long)t * 16;
  G[0] = gx; G[1] = cx; G[2]  = ax; G[3]  = tx;
  G[4] = gy; G[5] = cy; G[6]  = ay; G[7]  = ty;
  G[8] = gz; G[9] = cz; G[10] = az; G[11] = tz;
  G[12] = 0.f; G[13] = 0.f; G[14] = 0.f; G[15] = 1.f;
  out[NT * 16 + t] = 1.f / (1.f + expf(-segr[t]));
  float* xo = out + NT * 16 + NT + (long long)t * 3;
  xo[0] = pt[0]; xo[1] = pt[1]; xo[2] = pt[2];
}

// ================================================================ host
struct Lay { const float *W, *b, *g, *be; int cin, cout, cinPad; _Float16* Wh; };
struct Head2 { const float *W, *b; int cout; };

extern "C" void kernel_launch(void* const* d_in, const int* in_sizes, int n_in,
                              void* d_out, int out_size, void* d_ws, size_t ws_size,
                              hipStream_t stream) {
  (void)in_sizes; (void)n_in; (void)out_size; (void)ws_size;
  const int Bc = 2, N0 = 20000, S1 = 2048, S2 = 512, S3 = 128;

  // ---- parse params (jax pytree order: dict keys sorted; 'params' < 'xyz_pc')
  int pi = 0;
  auto FPp = [&](int i) { return (const float*)d_in[i]; };
  Lay all[48]; int nl = 0;
  auto addL = [&](int cin, int cout) -> Lay* {
    Lay L; L.W = FPp(pi); L.b = FPp(pi + 1); L.g = FPp(pi + 2); L.be = FPp(pi + 3);
    pi += 4; L.cin = cin; L.cout = cout; L.cinPad = (cin + 31) & ~31; L.Wh = nullptr;
    all[nl] = L; return &all[nl++];
  };
  auto addHead = [&](int cin, int cout, Lay** l1, Head2* h2) {
    *l1 = addL(cin, 128);
    h2->W = FPp(pi); h2->b = FPp(pi + 1); pi += 2; h2->cout = cout;
  };
  auto addMLP3 = [&](int cin0, const int* couts, Lay** dst) {
    int cin = cin0;
    for (int i = 0; i < 3; ++i) { dst[i] = addL(cin, couts[i]); cin = couts[i]; }
  };

  Lay *app1, *dir1, *off1, *seg1; Head2 app2, dir2, off2, seg2;
  Lay *fp1s[2], *fp2s[2], *fp3s[3];
  Lay *sa1s[3][3], *sa2s[3][3], *sa3s[3][3], *sa4s[3];

  addHead(256, 3, &app1, &app2);                         // 'app'
  addHead(256, 3, &dir1, &dir2);                         // 'dir'
  fp1s[0] = addL(448, 256);  fp1s[1] = addL(256, 256);   // 'fp1'
  fp2s[0] = addL(768, 256);  fp2s[1] = addL(256, 128);   // 'fp2'
  fp3s[0] = addL(1664, 128); fp3s[1] = addL(128, 128); fp3s[2] = addL(128, 128); // 'fp3'
  addHead(256, 10, &off1, &off2);                        // 'off'
  { int a[3] = {32, 32, 64}, b[3] = {64, 64, 128}, c[3] = {64, 96, 128};
    addMLP3(6, a, sa1s[0]); addMLP3(6, b, sa1s[1]); addMLP3(6, c, sa1s[2]); }   // 'sa1'
  { int a[3] = {64, 64, 128}, b[3] = {128, 128, 256}, c[3] = {128, 128, 256};
    addMLP3(323, a, sa2s[0]); addMLP3(323, b, sa2s[1]); addMLP3(323, c, sa2s[2]); } // 'sa2'
  { int a[3] = {64, 64, 128}, b[3] = {128, 128, 256}, c[3] = {128, 256, 256};
    addMLP3(643, a, sa3s[0]); addMLP3(643, b, sa3s[1]); addMLP3(643, c, sa3s[2]); } // 'sa3'
  { int a[3] = {256, 512, 1024}; addMLP3(643, a, sa4s); }                        // 'sa4'
  addHead(256, 1, &seg1, &seg2);                         // 'seg'
  const float* xyz = FPp(pi);                            // 'xyz_pc'

  // ---- workspace bump allocator
  char* basep = (char*)d_ws; size_t cur = 0;
  auto alloc = [&](size_t bytes) -> void* {
    cur = (cur + 255) & ~(size_t)255;
    void* p = basep + cur; cur += bytes; return p;
  };

  for (int i = 0; i < nl; ++i) {
    Lay& L = all[i];
    L.Wh = (_Float16*)alloc((size_t)L.cinPad * L.cout * sizeof(_Float16));
    int tot = L.cinPad * L.cout;
    k_w2h<<<(tot + 255) / 256, 256, 0, stream>>>(L.W, L.Wh, L.cin, L.cinPad, L.cout);
  }

  float* dist  = (float*)alloc((size_t)Bc * N0 * 4);
  int* fidx1   = (int*)alloc((size_t)Bc * S1 * 4);
  int* fidx2   = (int*)alloc((size_t)Bc * S2 * 4);
  int* fidx3   = (int*)alloc((size_t)Bc * S3 * 4);
  float* nx1   = (float*)alloc((size_t)Bc * S1 * 3 * 4);
  float* nx2   = (float*)alloc((size_t)Bc * S2 * 3 * 4);
  float* nx3   = (float*)alloc((size_t)Bc * S3 * 3 * 4);
  int* gidx    = (int*)alloc((size_t)Bc * S1 * 128 * 4);
  _Float16* X  = (_Float16*)alloc((size_t)131072 * 352 * 2);   // max rows*cinPad
  _Float16* Ya = (_Float16*)alloc((size_t)524288 * 128 * 2);   // max rows*cout
  _Float16* Yb = (_Float16*)alloc((size_t)524288 * 128 * 2);
  float* l1p   = (float*)alloc((size_t)Bc * S1 * 320 * 4);
  float* l2p   = (float*)alloc((size_t)Bc * S2 * 640 * 4);
  float* l3p   = (float*)alloc((size_t)Bc * S3 * 640 * 4);
  float* l4p   = (float*)alloc((size_t)Bc * 1024 * 4);
  float* interp= (float*)alloc((size_t)Bc * S1 * 1024 * 4);
  float* l3p2  = (float*)alloc((size_t)Bc * S3 * 128 * 4);
  float* l2p2  = (float*)alloc((size_t)Bc * S2 * 128 * 4);
  _Float16* featsH = (_Float16*)alloc((size_t)Bc * S1 * 256 * 2);
  float* gdr   = (float*)alloc((size_t)Bc * S1 * 3 * 4);
  float* adr   = (float*)alloc((size_t)Bc * S1 * 3 * 4);
  float* wlog  = (float*)alloc((size_t)Bc * S1 * 10 * 4);
  float* segr  = (float*)alloc((size_t)Bc * S1 * 4);

  auto gemm = [&](const _Float16* Xin, Lay* L, _Float16* Yh, float* Yf, int M) {
    dim3 grid((M + 127) / 128, L->cout / 32);
    k_wmma_mlp<<<grid, 256, 0, stream>>>(Xin, L->Wh, L->b, L->g, L->be, Yh, Yf,
                                         M, L->cinPad, L->cout);
  };
  auto run_stack = [&](Lay** Ls, int n, const _Float16* X0, int M,
                       _Float16* YhLast, float* YfLast) {
    const _Float16* curp = X0;
    for (int i = 0; i < n; ++i) {
      _Float16* Yh; float* Yf = nullptr;
      if (i == n - 1) { Yh = YfLast ? nullptr : YhLast; Yf = YfLast; }
      else Yh = (curp == Ya) ? Yb : Ya;
      gemm(curp, Ls[i], Yh, Yf, M);
      curp = Yh;
    }
  };
  auto sa_branch = [&](const float* pxyz, const float* nxyz, const float* feats, int D,
                       int S, int N, float radius, int K, Lay** Ls,
                       float* outFeat, int Ctot, int coff) {
    int M = Bc * S * K;
    int tq = Bc * S;
    k_ballquery<<<(tq + 255) / 256, 256, 0, stream>>>(nxyz, pxyz, gidx, Bc, S, N,
                                                      radius * radius, K);
    long long tg = (long long)M;
    k_group<<<(int)((tg + 255) / 256), 256, 0, stream>>>(pxyz, nxyz, feats, gidx, X,
                                                         Bc, S, N, K, D, Ls[0]->cinPad);
    run_stack(Ls, 3, X, M, Ya, nullptr);   // stack ends in Ya (X->Ya->Yb->Ya)
    long long tp = (long long)Bc * S * Ls[2]->cout;
    k_maxpool<<<(int)((tp + 255) / 256), 256, 0, stream>>>(Ya, outFeat, Bc, S, K,
                                                           Ls[2]->cout, Ctot, coff);
  };

  // ---- SA1
  k_fps<<<Bc, 256, 0, stream>>>(xyz, dist, fidx1, N0, S1);
  k_gather3<<<(Bc * S1 + 255) / 256, 256, 0, stream>>>(xyz, fidx1, nx1, Bc, S1, N0);
  sa_branch(xyz, nx1, xyz, 3, S1, N0, 0.02f, 32,  sa1s[0], l1p, 320, 0);
  sa_branch(xyz, nx1, xyz, 3, S1, N0, 0.04f, 64,  sa1s[1], l1p, 320, 64);
  sa_branch(xyz, nx1, xyz, 3, S1, N0, 0.08f, 128, sa1s[2], l1p, 320, 192);

  // ---- SA2
  k_fps<<<Bc, 256, 0, stream>>>(nx1, dist, fidx2, S1, S2);
  k_gather3<<<(Bc * S2 + 255) / 256, 256, 0, stream>>>(nx1, fidx2, nx2, Bc, S2, S1);
  sa_branch(nx1, nx2, l1p, 320, S2, S1, 0.04f, 64,  sa2s[0], l2p, 640, 0);
  sa_branch(nx1, nx2, l1p, 320, S2, S1, 0.08f, 64,  sa2s[1], l2p, 640, 128);
  sa_branch(nx1, nx2, l1p, 320, S2, S1, 0.16f, 128, sa2s[2], l2p, 640, 384);

  // ---- SA3
  k_fps<<<Bc, 256, 0, stream>>>(nx2, dist, fidx3, S2, S3);
  k_gather3<<<(Bc * S3 + 255) / 256, 256, 0, stream>>>(nx2, fidx3, nx3, Bc, S3, S2);
  sa_branch(nx2, nx3, l2p, 640, S3, S2, 0.08f, 64,  sa3s[0], l3p, 640, 0);
  sa_branch(nx2, nx3, l2p, 640, S3, S2, 0.16f, 64,  sa3s[1], l3p, 640, 128);
  sa_branch(nx2, nx3, l2p, 640, S3, S2, 0.32f, 128, sa3s[2], l3p, 640, 384);

  // ---- SA4 (group-all: concat [xyz, feats])
  {
    long long rows = (long long)Bc * S3;
    long long tc = rows * sa4s[0]->cinPad;
    k_concat2h<<<(int)((tc + 255) / 256), 256, 0, stream>>>(nx3, l3p, X, rows, 3, 640,
                                                            sa4s[0]->cinPad);
    run_stack(sa4s, 3, X, (int)rows, Ya, nullptr);
    long long tp = (long long)Bc * 1024;
    k_maxpool<<<(int)((tp + 255) / 256), 256, 0, stream>>>(Ya, l4p, Bc, 1, S3, 1024,
                                                           1024, 0);
  }

  // ---- FP3 (broadcast l4p)
  {
    long long tb = (long long)Bc * S3 * 1024;
    k_broadcast<<<(int)((tb + 255) / 256), 256, 0, stream>>>(l4p, interp, Bc, S3, 1024);
    long long rows = (long long)Bc * S3;
    long long tc = rows * fp3s[0]->cinPad;
    k_concat2h<<<(int)((tc + 255) / 256), 256, 0, stream>>>(l3p, interp, X, rows, 640,
                                                            1024, fp3s[0]->cinPad);
    run_stack(fp3s, 3, X, (int)rows, nullptr, l3p2);
  }
  // ---- FP2
  {
    k_interp3<<<(Bc * S2 + 255) / 256, 256, 0, stream>>>(nx2, nx3, l3p2, interp,
                                                         Bc, S2, S3, 128);
    long long rows = (long long)Bc * S2;
    long long tc = rows * fp2s[0]->cinPad;
    k_concat2h<<<(int)((tc + 255) / 256), 256, 0, stream>>>(l2p, interp, X, rows, 640,
                                                            128, fp2s[0]->cinPad);
    run_stack(fp2s, 2, X, (int)rows, nullptr, l2p2);
  }
  // ---- FP1 -> feats (fp16 for heads)
  {
    k_interp3<<<(Bc * S1 + 255) / 256, 256, 0, stream>>>(nx1, nx2, l2p2, interp,
                                                         Bc, S1, S2, 128);
    long long rows = (long long)Bc * S1;
    long long tc = rows * fp1s[0]->cinPad;
    k_concat2h<<<(int)((tc + 255) / 256), 256, 0, stream>>>(l1p, interp, X, rows, 320,
                                                            128, fp1s[0]->cinPad);
    run_stack(fp1s, 2, X, (int)rows, featsH, nullptr);
  }

  // ---- heads
  auto head = [&](Lay* L1, Head2& H2, float* outbuf) {
    gemm(featsH, L1, Ya, nullptr, Bc * S1);
    int tot = Bc * S1 * H2.cout;
    k_linear2<<<(tot + 255) / 256, 256, 0, stream>>>(Ya, H2.W, H2.b, outbuf,
                                                     Bc * S1, H2.cout);
  };
  head(dir1, dir2, gdr);
  head(app1, app2, adr);
  head(off1, off2, wlog);
  head(seg1, seg2, segr);

  k_post<<<(Bc * S1 + 255) / 256, 256, 0, stream>>>(gdr, adr, wlog, segr, nx1,
                                                    (float*)d_out, Bc, S1);
}
